// PyGMOFuseMoE_80393197846542
// MI455X (gfx1250) — matrix operand order repaired and checked by reading, not compiled, not run on hardware
//
#include <hip/hip_runtime.h>
#include <hip/hip_bf16.h>

// ---------------- problem sizes ----------------
constexpr int B_   = 8192;
constexpr int IN_  = 1024;
constexpr int H_   = 4096;
constexpr int OUT_ = 1024;
constexpr int E_   = 8;
constexpr int HG_  = 2048;

constexpr int TM  = 32;    // batch rows per workgroup
constexpr int HCH = 512;   // H-chunk kept in LDS between the two expert GEMMs
constexpr int KC  = 32;    // WMMA K per step (bf16)
constexpr int GNC = 256;   // gating N-chunk

typedef __attribute__((ext_vector_type(16))) __bf16 bf16x16;
typedef __attribute__((ext_vector_type(8)))  __bf16 bf16x8;
typedef __attribute__((ext_vector_type(4)))  __bf16 bf16x4;
typedef __attribute__((ext_vector_type(8)))  float  f32x8;
typedef __attribute__((ext_vector_type(4)))  float  f32x4;

__device__ __forceinline__ f32x8 zero8() {
  f32x8 z;
#pragma unroll
  for (int i = 0; i < 8; ++i) z[i] = 0.0f;
  return z;
}

__device__ __forceinline__ bf16x16 cat16(bf16x8 lo, bf16x8 hi) {
  return __builtin_shufflevector(lo, hi, 0,1,2,3,4,5,6,7,8,9,10,11,12,13,14,15);
}

__device__ __forceinline__ f32x8 wmma_bf16(bf16x16 a, bf16x16 b, f32x8 c) {
  // D = A(16x32) x B(32x16) + C, f32 accumulate
  return __builtin_amdgcn_wmma_f32_16x16x32_bf16(false, a, false, b, (short)0, c,
                                                 false, false);
}

// A fragment (16x32 bf16) from a row-major source (row stride in elements).
// ISA 7.12.2: lanes 0-15 hold M=lane, K {0..7,16..23}; lanes 16-31 hold K {8..15,24..31}.
__device__ __forceinline__ bf16x16 load_a(const __bf16* base, int stride, int lid) {
  const int m  = lid & 15;
  const int k0 = (lid < 16) ? 0 : 8;
  const __bf16* p = base + (size_t)m * stride + k0;
  bf16x8 lo = *(const bf16x8*)(p);
  bf16x8 hi = *(const bf16x8*)(p + 16);
  return cat16(lo, hi);
}

// B fragment (32x16 bf16) from an LDS tile stored transposed as [n][KC]:
// lanes 0-15 -> N=lane, K {0..7,16..23} ; lanes 16-31 -> N=lane-16, K {8..15,24..31}.
__device__ __forceinline__ bf16x16 load_b(const __bf16* wt, int ncol, int lid) {
  const int n  = ncol + (lid & 15);
  const int k0 = (lid < 16) ? 0 : 8;
  const __bf16* p = wt + n * KC + k0;
  bf16x8 lo = *(const bf16x8*)(p);
  bf16x8 hi = *(const bf16x8*)(p + 16);
  return cat16(lo, hi);
}

// ---- software-pipelined weight-tile staging (KC x NC f32 -> bf16, transposed) ----
template<int NC>
struct SReg { f32x4 v[(KC * NC) / 1024]; };   // per-thread staged registers

template<int NC>
__device__ __forceinline__ void stage_load(const float* __restrict__ src, int nstride,
                                           SReg<NC>& r, int tid) {
  constexpr int PT = (KC * NC) / 1024;
#pragma unroll
  for (int j = 0; j < PT; ++j) {
    const int g  = tid + j * 256;
    const int k  = g / (NC / 4);
    const int n4 = (g % (NC / 4)) * 4;
    const float* p = src + (size_t)k * nstride + n4;
    __builtin_prefetch(p + (size_t)(2 * KC) * nstride, 0, 1);  // 2 tiles ahead
    r.v[j] = *(const f32x4*)(p);
  }
}

template<int NC>
__device__ __forceinline__ void stage_store(const SReg<NC>& r, __bf16* dst, int tid) {
  constexpr int PT = (KC * NC) / 1024;
#pragma unroll
  for (int j = 0; j < PT; ++j) {
    const int g  = tid + j * 256;
    const int k  = g / (NC / 4);
    const int n4 = (g % (NC / 4)) * 4;
    dst[(n4 + 0) * KC + k] = (__bf16)r.v[j][0];
    dst[(n4 + 1) * KC + k] = (__bf16)r.v[j][1];
    dst[(n4 + 2) * KC + k] = (__bf16)r.v[j][2];
    dst[(n4 + 3) * KC + k] = (__bf16)r.v[j][3];
  }
}

// ---------------- x: f32 -> bf16 ----------------
__global__ __launch_bounds__(256) void cvt_bf16_kernel(const float* __restrict__ x,
                                                       __bf16* __restrict__ xbf,
                                                       int n) {
  const int i = (blockIdx.x * 256 + threadIdx.x) * 4;
  if (i + 3 < n) {
    f32x4 v = *(const f32x4*)(x + i);
    bf16x4 o;
    o[0] = (__bf16)v[0]; o[1] = (__bf16)v[1];
    o[2] = (__bf16)v[2]; o[3] = (__bf16)v[3];
    *(bf16x4*)(xbf + i) = o;
  }
}

// ---------------- gating: softmax(relu(x gw1 + b1) gw2 + b2) ----------------
__global__ __launch_bounds__(256) void gating_kernel(
    const __bf16* __restrict__ xbf, const float* __restrict__ gw1,
    const float* __restrict__ gb1, const float* __restrict__ gw2,
    const float* __restrict__ gb2, float* __restrict__ gates) {
  __shared__ __attribute__((aligned(16))) __bf16 wt[2][GNC * KC];  // 2 x 16 KB ping-pong
  __shared__ __attribute__((aligned(16))) __bf16 ghs[TM * GNC];    // 16 KB
  __shared__ float lr[TM * E_];                                    // 1 KB

  const int tid = threadIdx.x;
  const int lid = tid & 31;
  const int w   = tid >> 5;
  const int mt  = w & 1;
  const int wn  = w >> 1;          // 0..3
  const int rb  = blockIdx.x * TM;
  const int trow = tid >> 3;       // logits: one (row,e) pair per thread
  const int te   = tid & 7;

  float lacc = gb2[te];

  for (int hc = 0; hc < HG_; hc += GNC) {
    f32x8 acc[4];
#pragma unroll
    for (int t = 0; t < 4; ++t) acc[t] = zero8();

    SReg<GNC> sreg;
    stage_load<GNC>(gw1 + hc, HG_, sreg, tid);        // kb = 0
    int cur = 0;
    for (int kb = 0; kb < IN_; kb += KC) {
      stage_store<GNC>(sreg, &wt[cur][0], tid);
      if (kb + KC < IN_)
        stage_load<GNC>(gw1 + (size_t)(kb + KC) * HG_ + hc, HG_, sreg, tid);
      __syncthreads();
      bf16x16 a = load_a(xbf + (size_t)(rb + mt * 16) * IN_ + kb, IN_, lid);
      bf16x16 bfr[4];
#pragma unroll
      for (int t = 0; t < 4; ++t) bfr[t] = load_b(&wt[cur][0], (wn * 4 + t) * 16, lid);
#pragma unroll
      for (int t = 0; t < 4; ++t) acc[t] = wmma_bf16(a, bfr[t], acc[t]);
      cur ^= 1;
    }
    __syncthreads();
    // relu(+bias) -> LDS gh chunk (bf16)
#pragma unroll
    for (int t = 0; t < 4; ++t) {
      const int nl  = (wn * 4 + t) * 16 + (lid & 15);
      const float b1 = gb1[hc + nl];
#pragma unroll
      for (int r = 0; r < 8; ++r) {
        const int m = mt * 16 + r + ((lid < 16) ? 0 : 8);
        ghs[m * GNC + nl] = (__bf16)fmaxf(acc[t][r] + b1, 0.0f);
      }
    }
    __syncthreads();
    // logits partial reduction over this chunk
    for (int k = 0; k < GNC; ++k)
      lacc += (float)ghs[trow * GNC + k] * gw2[(size_t)(hc + k) * E_ + te];
  }
  __syncthreads();
  lr[trow * E_ + te] = lacc;
  __syncthreads();
  float mx = -3.0e38f;
#pragma unroll
  for (int j = 0; j < E_; ++j) mx = fmaxf(mx, lr[trow * E_ + j]);
  float s = 0.0f;
#pragma unroll
  for (int j = 0; j < E_; ++j) s += __expf(lr[trow * E_ + j] - mx);
  gates[(size_t)(rb + trow) * E_ + te] = __expf(lacc - mx) / s;
}

// ------------- fused experts: out = sum_e (g_be*relu(xW1+b1)) W2 + (sum_e g_be b2) -
__global__ __launch_bounds__(256) void expert_kernel(
    const __bf16* __restrict__ xbf, const float* __restrict__ ew1,
    const float* __restrict__ eb1, const float* __restrict__ ew2,
    const float* __restrict__ eb2, const float* __restrict__ gates,
    float* __restrict__ out) {
  __shared__ __attribute__((aligned(16))) __bf16 wt[2][HCH * KC];  // 2 x 32 KB ping-pong
  __shared__ __attribute__((aligned(16))) __bf16 hbuf[TM * HCH];   // 32 KB

  const int tid = threadIdx.x;
  const int lid = tid & 31;
  const int w   = tid >> 5;
  const int mt  = w & 1;
  const int wn  = w >> 1;          // 0..3
  const int rb  = blockIdx.x * TM;

  f32x8 acc2[16];                  // persistent 32x1024 output accumulators
#pragma unroll
  for (int i = 0; i < 16; ++i) acc2[i] = zero8();

  for (int e = 0; e < E_; ++e) {
    const float* w1 = ew1 + (size_t)e * IN_ * H_;
    const float* w2 = ew2 + (size_t)e * H_ * OUT_;
    float gv[8];
#pragma unroll
    for (int r = 0; r < 8; ++r)
      gv[r] = gates[(size_t)(rb + mt * 16 + r + ((lid < 16) ? 0 : 8)) * E_ + e];

    for (int hc = 0; hc < H_; hc += HCH) {
      // ---- GEMM1 chunk: h[0:32, hc:hc+HCH], double-buffered ----
      f32x8 acc1[8];
#pragma unroll
      for (int t = 0; t < 8; ++t) acc1[t] = zero8();
      {
        SReg<HCH> sreg;
        stage_load<HCH>(w1 + hc, H_, sreg, tid);      // kb = 0
        int cur = 0;
        for (int kb = 0; kb < IN_; kb += KC) {
          stage_store<HCH>(sreg, &wt[cur][0], tid);
          if (kb + KC < IN_)
            stage_load<HCH>(w1 + (size_t)(kb + KC) * H_ + hc, H_, sreg, tid);
          __syncthreads();
          bf16x16 a = load_a(xbf + (size_t)(rb + mt * 16) * IN_ + kb, IN_, lid);
          bf16x16 bfr[4];
#pragma unroll
          for (int t = 0; t < 4; ++t) bfr[t] = load_b(&wt[cur][0], (wn * 8 + t) * 16, lid);
#pragma unroll
          for (int t = 0; t < 4; ++t) acc1[t] = wmma_bf16(a, bfr[t], acc1[t]);
#pragma unroll
          for (int t = 0; t < 4; ++t) bfr[t] = load_b(&wt[cur][0], (wn * 8 + t + 4) * 16, lid);
#pragma unroll
          for (int t = 0; t < 4; ++t) acc1[t + 4] = wmma_bf16(a, bfr[t], acc1[t + 4]);
          cur ^= 1;
        }
      }
      __syncthreads();
      // epilogue: gate * relu(h + b1) -> LDS (bf16)
#pragma unroll
      for (int t = 0; t < 8; ++t) {
        const int nl  = (wn * 8 + t) * 16 + (lid & 15);
        const float b1 = eb1[(size_t)e * H_ + hc + nl];
#pragma unroll
        for (int r = 0; r < 8; ++r) {
          const int m = mt * 16 + r + ((lid < 16) ? 0 : 8);
          hbuf[m * HCH + nl] = (__bf16)(fmaxf(acc1[t][r] + b1, 0.0f) * gv[r]);
        }
      }
      __syncthreads();
      // ---- GEMM2 partial: acc2 += hbuf @ W2[hc:hc+HCH, :], double-buffered ----
      {
        SReg<512> sreg;
        stage_load<512>(w2 + (size_t)hc * OUT_, OUT_, sreg, tid);   // step 0
        int cur = 0;
        for (int s = 0; s < 32; ++s) {                // s = (p<<4) | (kk/KC)
          const int p  = s >> 4;
          const int kk = (s & 15) * KC;
          stage_store<512>(sreg, &wt[cur][0], tid);
          if (s + 1 < 32) {
            const int ns  = s + 1;
            const int np  = ns >> 4;
            const int nkk = (ns & 15) * KC;
            stage_load<512>(w2 + (size_t)(hc + nkk) * OUT_ + np * 512, OUT_, sreg, tid);
          }
          __syncthreads();
          bf16x16 a = load_a(hbuf + mt * 16 * HCH + kk, HCH, lid);
          bf16x16 bfr[4];
#pragma unroll
          for (int t = 0; t < 4; ++t) bfr[t] = load_b(&wt[cur][0], (wn * 8 + t) * 16, lid);
#pragma unroll
          for (int t = 0; t < 4; ++t)
            acc2[p * 8 + t] = wmma_bf16(a, bfr[t], acc2[p * 8 + t]);
#pragma unroll
          for (int t = 0; t < 4; ++t) bfr[t] = load_b(&wt[cur][0], (wn * 8 + t + 4) * 16, lid);
#pragma unroll
          for (int t = 0; t < 4; ++t)
            acc2[p * 8 + t + 4] = wmma_bf16(a, bfr[t], acc2[p * 8 + t + 4]);
          cur ^= 1;
        }
      }
      __syncthreads();   // wt / hbuf safe to overwrite next chunk
    }
  }
  // final epilogue: add sum_e gates*b2, plain deterministic stores
#pragma unroll
  for (int p = 0; p < 2; ++p)
#pragma unroll
    for (int t = 0; t < 8; ++t) {
      const int col = p * 512 + (wn * 8 + t) * 16 + (lid & 15);
      float bv[E_];
#pragma unroll
      for (int e = 0; e < E_; ++e) bv[e] = eb2[(size_t)e * OUT_ + col];
#pragma unroll
      for (int r = 0; r < 8; ++r) {
        const int row = rb + mt * 16 + r + ((lid < 16) ? 0 : 8);
        float bias = 0.0f;
#pragma unroll
        for (int e = 0; e < E_; ++e) bias += gates[(size_t)row * E_ + e] * bv[e];
        out[(size_t)row * OUT_ + col] = acc2[p * 8 + t][r] + bias;
      }
    }
}

extern "C" void kernel_launch(void* const* d_in, const int* in_sizes, int n_in,
                              void* d_out, int out_size, void* d_ws, size_t ws_size,
                              hipStream_t stream) {
  const float* x   = (const float*)d_in[0];
  const float* gw1 = (const float*)d_in[1];
  const float* gb1 = (const float*)d_in[2];
  const float* gw2 = (const float*)d_in[3];
  const float* gb2 = (const float*)d_in[4];
  const float* ew1 = (const float*)d_in[5];
  const float* eb1 = (const float*)d_in[6];
  const float* ew2 = (const float*)d_in[7];
  const float* eb2 = (const float*)d_in[8];
  float* out = (float*)d_out;

  // workspace layout: [ x_bf16 : 16 MB ][ gates : 256 KB ]
  __bf16* xbf  = (__bf16*)d_ws;
  float* gates = (float*)((char*)d_ws + (size_t)B_ * IN_ * sizeof(__bf16));

  cvt_bf16_kernel<<<(B_ * IN_) / (4 * 256), 256, 0, stream>>>(x, xbf, B_ * IN_);
  gating_kernel<<<B_ / TM, 256, 0, stream>>>(xbf, gw1, gb1, gw2, gb2, gates);
  expert_kernel<<<B_ / TM, 256, 0, stream>>>(xbf, ew1, eb1, ew2, eb2, gates, out);
}
